// TransformerSelfAttention_38173669327385
// MI455X (gfx1250) — matrix operand order repaired and checked by reading, not compiled, not run on hardware
//
#include <hip/hip_runtime.h>

typedef __attribute__((ext_vector_type(16))) __bf16 v16bf;
typedef __attribute__((ext_vector_type(8)))  __bf16 v8bf;
typedef __attribute__((ext_vector_type(8)))  float  v8f;
typedef __attribute__((ext_vector_type(4)))  unsigned int v4u;
typedef __attribute__((ext_vector_type(8)))  int v8i;
typedef __attribute__((ext_vector_type(4)))  int v4i;

#define WMMA_BF16(A, B, C) \
  __builtin_amdgcn_wmma_f32_16x16x32_bf16(false, (A), false, (B), (short)0, (C), false, false)

#define NB    2
#define NS_   2048
#define ND    768
#define NH    12
#define NHD   64
#define NPAST 2048
#define NMAX  4096   // PAST + S

#if defined(__has_builtin)
#if __has_builtin(__builtin_amdgcn_tensor_load_to_lds) && \
    __has_builtin(__builtin_amdgcn_s_wait_tensorcnt)
#define HAVE_TDM 1
#endif
#endif

__device__ __forceinline__ v8bf pack_f4x2(float4 a, float4 b) {
  v8bf r;
  r[0] = (__bf16)a.x; r[1] = (__bf16)a.y; r[2] = (__bf16)a.z; r[3] = (__bf16)a.w;
  r[4] = (__bf16)b.x; r[5] = (__bf16)b.y; r[6] = (__bf16)b.z; r[7] = (__bf16)b.w;
  return r;
}
__device__ __forceinline__ v16bf cat8(v8bf lo, v8bf hi) {
  v16bf r;
#pragma unroll
  for (int j = 0; j < 8; ++j) { r[j] = lo[j]; r[j + 8] = hi[j]; }
  return r;
}

#ifdef HAVE_TDM
// Tensor DMA: 2-D bf16 tile (tile0 contiguous elems x tile1 rows, row stride
// stride0 elems) from global -> LDS, packed contiguously. D# per ISA ch.8.
// This toolchain's builtin takes 6 args (groups 2/3 as int32x4, plus an
// int32x8 trailing group) — pass zeros for the unused groups (2-D tensor).
__device__ __forceinline__ void tdm_load_2d(const void* lds_ptr, const void* gptr,
                                            unsigned tile0, unsigned tile1,
                                            unsigned tensor0, unsigned tensor1,
                                            unsigned stride0) {
  const unsigned long long ga = (unsigned long long)gptr;
  v4u g0;
  g0[0] = 1u;                                               // count=1 (valid), user mode
  g0[1] = (unsigned)(unsigned long long)lds_ptr;            // lds_addr (bytes)
  g0[2] = (unsigned)ga;                                     // global_addr[31:0]
  g0[3] = (unsigned)((ga >> 32) & 0x1ffffffull) | (2u << 30);  // addr[56:32] | type=2
  v8i g1;
  g1[0] = 1 << 16;                                          // data_size=1 -> 2 bytes
  g1[1] = (int)((tensor0 & 0xffffu) << 16);                 // tensor_dim0[15:0]
  g1[2] = (int)(((tensor0 >> 16) & 0xffffu) | ((tensor1 & 0xffffu) << 16));
  g1[3] = (int)(((tensor1 >> 16) & 0xffffu) | (tile0 << 16));  // tile_dim0
  g1[4] = (int)(tile1 & 0xffffu);                           // tile_dim1 (0=1D)
  g1[5] = (int)stride0;                                     // tensor_dim0_stride[31:0]
  g1[6] = 0;
  g1[7] = 0;
  v4i z4 = {0, 0, 0, 0};
  v8i z8 = {0, 0, 0, 0, 0, 0, 0, 0};
  __builtin_amdgcn_tensor_load_to_lds(g0, g1, z4, z4, z8, 0);
}
#endif

// ---------------------------------------------------------------------------
// Kernel 0: past KV -> present f32, plus bf16 mirrors: K [bh][key][d],
// V^T [bh][d][key]. V^T writes are contiguous via lane-coalesced strided reads.
// grid (NPAST/32, NB*NH), 256 threads, one 32x64 tile per block.
// ---------------------------------------------------------------------------
__global__ __launch_bounds__(256)
void past_kernel(const float* __restrict__ past, float* __restrict__ present,
                 __bf16* __restrict__ kws, __bf16* __restrict__ vtws) {
  const int key0 = blockIdx.x * 32;
  const int bh = blockIdx.y;
  const int t = threadIdx.x;
  // row-major role: 8 contiguous floats per thread
  const int row = t >> 3, seg = (t & 7) * 8;
  const float* pk = past + ((size_t)bh * NPAST + key0 + row) * NHD + seg;
  const float* pv = past + ((size_t)(NB * NH + bh) * NPAST + key0 + row) * NHD + seg;
  float4 a0 = *(const float4*)pk, a1 = *(const float4*)(pk + 4);
  float4 b0 = *(const float4*)pv, b1 = *(const float4*)(pv + 4);
  float* qk = present + ((size_t)bh * NMAX + key0 + row) * NHD + seg;
  float* qv = present + ((size_t)(NB * NH + bh) * NMAX + key0 + row) * NHD + seg;
  *(float4*)qk = a0; *(float4*)(qk + 4) = a1;
  *(float4*)qv = b0; *(float4*)(qv + 4) = b1;
  *(v8bf*)&kws[((size_t)bh * NMAX + key0 + row) * NHD + seg] = pack_f4x2(a0, a1);
  // V^T role: fixed d, 8 consecutive keys (reads lane-coalesced, write contiguous)
  const int d = t & 63, rh = (t >> 6) * 8;
  const float* ps = past + ((size_t)(NB * NH + bh) * NPAST + key0 + rh) * NHD + d;
  v8bf o;
#pragma unroll
  for (int i = 0; i < 8; ++i) o[i] = (__bf16)ps[(size_t)i * NHD];
  *(v8bf*)&vtws[((size_t)bh * NHD + d) * NMAX + key0 + rh] = o;
}

// ---------------------------------------------------------------------------
// Kernel 1: C^T = W^T X^T (+bias). mode 0: Q->qws bf16. mode 1: K->present f32
// + kws bf16. mode 2: V->present f32 + vtws bf16 (transposed scatter).
// ---------------------------------------------------------------------------
__global__ __launch_bounds__(256)
void qkv_kernel(const float* __restrict__ X, const float* __restrict__ W,
                const float* __restrict__ bias, __bf16* __restrict__ qws,
                float* __restrict__ present, __bf16* __restrict__ kws,
                __bf16* __restrict__ vtws, int mode) {
  __shared__ alignas(128) __bf16 xLds[128 * 32];  // [row][k]
  __shared__ alignas(128) __bf16 wLds[128 * 32];  // [col][k] (transposed)
  const int tid  = threadIdx.x;
  const int lane = tid & 31, wave = tid >> 5;
  const int half = lane >> 4, l15 = lane & 15;
  const int r0 = blockIdx.x * 128;
  const int c0 = blockIdx.y * 128;
  const int rg = wave >> 1, cg = wave & 1;

  const int sxrow = tid >> 1, sxseg = tid & 1;
  const float* xp = X + (size_t)(r0 + sxrow) * ND + sxseg * 16;
  const int swcol = tid & 127, swkh = tid >> 7;
  const float* wp = W + (size_t)(swkh * 16) * ND + c0 + swcol;

  v8f acc[4][2] = {};

  for (int kc = 0; kc < ND / 32; ++kc) {
    __syncthreads();
    float4 x0 = *(const float4*)(xp + 0);
    float4 x1 = *(const float4*)(xp + 4);
    float4 x2 = *(const float4*)(xp + 8);
    float4 x3 = *(const float4*)(xp + 12);
    *(v8bf*)&xLds[sxrow * 32 + sxseg * 16]     = pack_f4x2(x0, x1);
    *(v8bf*)&xLds[sxrow * 32 + sxseg * 16 + 8] = pack_f4x2(x2, x3);
    float wv[16];
#pragma unroll
    for (int i = 0; i < 16; ++i) wv[i] = wp[(size_t)i * ND];
    {
      v8bf a, b;
#pragma unroll
      for (int j = 0; j < 8; ++j) { a[j] = (__bf16)wv[j]; b[j] = (__bf16)wv[j + 8]; }
      *(v8bf*)&wLds[swcol * 32 + swkh * 16]     = a;
      *(v8bf*)&wLds[swcol * 32 + swkh * 16 + 8] = b;
    }
    __syncthreads();
    xp += 32;
    wp += (size_t)32 * ND;

    v16bf aw[4];
#pragma unroll
    for (int ct = 0; ct < 4; ++ct) {
      const int col = cg * 64 + ct * 16 + l15;
      aw[ct] = *(const v16bf*)&wLds[col * 32 + half * 16];
    }
    v16bf bx[2];
#pragma unroll
    for (int mt = 0; mt < 2; ++mt) {
      const int row = rg * 32 + mt * 16 + l15;
      bx[mt] = *(const v16bf*)&xLds[row * 32 + half * 16];
    }
#pragma unroll
    for (int ct = 0; ct < 4; ++ct)
#pragma unroll
      for (int mt = 0; mt < 2; ++mt)
        acc[ct][mt] = WMMA_BF16(aw[ct], bx[mt], acc[ct][mt]);
  }

#pragma unroll
  for (int ct = 0; ct < 4; ++ct) {
    const int colb = c0 + cg * 64 + ct * 16 + half * 8;
    const float4 bb0 = *(const float4*)&bias[colb];
    const float4 bb1 = *(const float4*)&bias[colb + 4];
    const int h = colb >> 6, hd0 = colb & 63;
#pragma unroll
    for (int mt = 0; mt < 2; ++mt) {
      const int row = r0 + rg * 32 + mt * 16 + l15;
      const int b = row >> 11, s = row & (NS_ - 1);
      const int bh = b * NH + h;
      float v[8];
      v[0] = acc[ct][mt][0] + bb0.x; v[1] = acc[ct][mt][1] + bb0.y;
      v[2] = acc[ct][mt][2] + bb0.z; v[3] = acc[ct][mt][3] + bb0.w;
      v[4] = acc[ct][mt][4] + bb1.x; v[5] = acc[ct][mt][5] + bb1.y;
      v[6] = acc[ct][mt][6] + bb1.z; v[7] = acc[ct][mt][7] + bb1.w;
      v8bf o;
#pragma unroll
      for (int j = 0; j < 8; ++j) o[j] = (__bf16)v[j];
      if (mode == 0) {
        *(v8bf*)&qws[((size_t)bh * NS_ + s) * NHD + hd0] = o;
      } else {
        const int kv = mode - 1;
        float* dst = present +
            (((size_t)(kv * NB * NH + bh)) * NMAX + (NPAST + s)) * NHD + hd0;
        *(float4*)dst       = make_float4(v[0], v[1], v[2], v[3]);
        *(float4*)(dst + 4) = make_float4(v[4], v[5], v[6], v[7]);
        if (mode == 1) {
          *(v8bf*)&kws[((size_t)bh * NMAX + NPAST + s) * NHD + hd0] = o;
        } else {
#pragma unroll
          for (int j = 0; j < 8; ++j)
            vtws[((size_t)bh * NHD + hd0 + j) * NMAX + NPAST + s] = o[j];
        }
      }
    }
  }
}

// ---------------------------------------------------------------------------
// Kernel 2: flash attention (transposed orientation), with TDM double-buffered
// bf16 tile staging: K tile = contiguous 4KB, V^T tile = 64 rows x 32 keys
// (stride NMAX). Wave 0 issues DMA; s_wait_tensorcnt keeps one pair in flight.
// ---------------------------------------------------------------------------
__global__ __launch_bounds__(256)
void attn_kernel(const __bf16* __restrict__ qws, const __bf16* __restrict__ kws,
                 const __bf16* __restrict__ vtws, float* __restrict__ ctx) {
  __shared__ alignas(128) __bf16 kLds[2][32 * 64];   // [key][d]
  __shared__ alignas(128) __bf16 vtLds[2][64 * 32];  // [d][key]

  const int tid  = threadIdx.x;
  const int lane = tid & 31, wave = tid >> 5;
  const int half = lane >> 4, l15 = lane & 15;
  const int q0wg = blockIdx.x * 128;
  const int bh = blockIdx.y;
  const int b = bh / NH, h = bh % NH;
  const __bf16* kwsb  = kws  + (size_t)bh * NMAX * NHD;
  const __bf16* vtwsb = vtws + (size_t)bh * NHD * NMAX;
  const int q0 = q0wg + wave * 16;

  const __bf16* qp = qws + (((size_t)bh) * NS_ + q0 + l15) * NHD;
  v16bf qb[2];
#pragma unroll
  for (int dc = 0; dc < 2; ++dc)
    qb[dc] = cat8(*(const v8bf*)(qp + dc * 32 + half * 8), *(const v8bf*)(qp + dc * 32 + 16 + half * 8));

  v8f acc[4] = {};
  float m = -1e30f, l = 0.0f;

  const int nkb = (NPAST + q0wg + 128) >> 5;

#ifdef HAVE_TDM
  if (tid < 32) {
    tdm_load_2d(&kLds[0][0], kwsb, 2048, 0, 2048, 1, 2048);
    tdm_load_2d(&vtLds[0][0], vtwsb, 32, 64, 32, 64, NMAX);
  }
#endif

  for (int kb = 0; kb < nkb; ++kb) {
    const int key0 = kb * 32;
    __syncthreads();  // all waves done reading the buffer the next DMA overwrites
#ifdef HAVE_TDM
    const int bufsel = kb & 1;
    if (tid < 32) {
      if (kb + 1 < nkb) {
        const int nk0 = (kb + 1) * 32;
        tdm_load_2d(&kLds[(kb + 1) & 1][0], kwsb + (size_t)nk0 * NHD,
                    2048, 0, 2048, 1, 2048);
        tdm_load_2d(&vtLds[(kb + 1) & 1][0], vtwsb + nk0, 32, 64, 32, 64, NMAX);
        __builtin_amdgcn_s_wait_tensorcnt((short)2);
      } else {
        __builtin_amdgcn_s_wait_tensorcnt((short)0);
      }
    }
    __syncthreads();
#else
    const int bufsel = 0;
    {
      const int row = tid >> 3, seg = (tid & 7) * 8;
      *(v8bf*)&kLds[0][row * 64 + seg] =
          *(const v8bf*)&kwsb[(size_t)(key0 + row) * NHD + seg];
      const int d = tid >> 2, ks = (tid & 3) * 8;
      *(v8bf*)&vtLds[0][d * 32 + ks] =
          *(const v8bf*)&vtwsb[(size_t)d * NMAX + key0 + ks];
    }
    __syncthreads();
#endif
    const __bf16* kT = &kLds[bufsel][0];
    const __bf16* vT = &vtLds[bufsel][0];

    // S^T tiles: element r -> key = key0 + kt*16 + half*8 + r, lane -> q
    v8f st[2];
#pragma unroll
    for (int kt = 0; kt < 2; ++kt) {
      v8f z = {};
#pragma unroll
      for (int dc = 0; dc < 2; ++dc) {
        v8bf lo = *(const v8bf*)&kT[(kt * 16 + l15) * 64 + dc * 32 + half * 8];
        v8bf hi = *(const v8bf*)&kT[(kt * 16 + l15) * 64 + dc * 32 + 16 + half * 8];
        z = WMMA_BF16(cat8(lo, hi), qb[dc], z);
      }
      st[kt] = z;
    }

    // causal mask + online softmax (q on lane axis -> scalar m/l per lane)
    const int kthr = NPAST + q0 + l15;
    float mloc = -1e30f;
#pragma unroll
    for (int kt = 0; kt < 2; ++kt)
#pragma unroll
      for (int r = 0; r < 8; ++r) {
        const int key = key0 + kt * 16 + half * 8 + r;
        float v = st[kt][r] * 0.125f;
        v = (key <= kthr) ? v : -1e10f;
        st[kt][r] = v;
        mloc = fmaxf(mloc, v);
      }
    mloc = fmaxf(mloc, __shfl_xor(mloc, 16, 32));
    const float mnew = fmaxf(m, mloc);
    const float scl = __expf(m - mnew);
    float ssum = 0.0f;
    v8f p[2];
#pragma unroll
    for (int kt = 0; kt < 2; ++kt)
#pragma unroll
      for (int r = 0; r < 8; ++r) {
        const float pv = __expf(st[kt][r] - mnew);
        p[kt][r] = pv;
        ssum += pv;
      }
    ssum += __shfl_xor(ssum, 16, 32);
    l = l * scl + ssum;
    m = mnew;
#pragma unroll
    for (int dt = 0; dt < 4; ++dt)
#pragma unroll
      for (int r = 0; r < 8; ++r) acc[dt][r] *= scl;

    // P^T B-operand built in registers from the lane's own p[0], p[1]; no exchange needed
    v8f mine = half ? p[1] : p[0];
    v8f xchs = half ? p[0] : p[1];
    v8f other;
#pragma unroll
    for (int r = 0; r < 8; ++r) other[r] = __shfl_xor(xchs[r], 16, 32);
    v16bf pb;
#pragma unroll
    for (int j = 0; j < 8; ++j) {
      pb[j]     = (__bf16)p[0][j];
      pb[j + 8] = (__bf16)p[1][j];
    }

    // ctx^T += V^T * P^T
#pragma unroll
    for (int dt = 0; dt < 4; ++dt) {
      v8bf lo = *(const v8bf*)&vT[(dt * 16 + l15) * 32 + half * 8];
      v8bf hi = *(const v8bf*)&vT[(dt * 16 + l15) * 32 + 16 + half * 8];
      acc[dt] = WMMA_BF16(cat8(lo, hi), pb, acc[dt]);
    }
  }

  const float inv = 1.0f / l;
  const int q = q0 + l15;
  float* op = ctx + ((size_t)b * NS_ + q) * ND + h * NHD;
#pragma unroll
  for (int dt = 0; dt < 4; ++dt) {
    const int d0 = dt * 16 + half * 8;
    *(float4*)(op + d0) =
        make_float4(acc[dt][0] * inv, acc[dt][1] * inv, acc[dt][2] * inv, acc[dt][3] * inv);
    *(float4*)(op + d0 + 4) =
        make_float4(acc[dt][4] * inv, acc[dt][5] * inv, acc[dt][6] * inv, acc[dt][7] * inv);
  }
}

// ---------------------------------------------------------------------------
extern "C" void kernel_launch(void* const* d_in, const int* in_sizes, int n_in,
                              void* d_out, int out_size, void* d_ws, size_t ws_size,
                              hipStream_t stream) {
  (void)in_sizes; (void)n_in; (void)out_size; (void)ws_size;
  const float* hidden = (const float*)d_in[0];
  const float* past   = (const float*)d_in[1];
  // d_in[2] = mask: analytic causal tril -> never read
  const float* Wq = (const float*)d_in[3];
  const float* bq = (const float*)d_in[4];
  const float* Wk = (const float*)d_in[5];
  const float* bk = (const float*)d_in[6];
  const float* Wv = (const float*)d_in[7];
  const float* bv = (const float*)d_in[8];

  float* out = (float*)d_out;
  float* ctx = out;                                    // (B, S, D)
  float* present = out + (size_t)NB * NS_ * ND;        // (2, B, H, 4096, 64)

  // workspace: Q bf16 (6.29MB) | K bf16 (12.58MB) | V^T bf16 (12.58MB)
  __bf16* qws  = (__bf16*)d_ws;
  __bf16* kws  = qws + (size_t)NB * NH * NS_ * NHD;
  __bf16* vtws = kws + (size_t)NB * NH * NMAX * NHD;

  dim3 g0(NPAST / 32, NB * NH);                        // (64, 24)
  past_kernel<<<g0, 256, 0, stream>>>(past, present, kws, vtws);

  dim3 g1(NB * NS_ / 128, ND / 128);                   // (32, 6)
  qkv_kernel<<<g1, 256, 0, stream>>>(hidden, Wq, bq, qws, present, kws, vtws, 0);
  qkv_kernel<<<g1, 256, 0, stream>>>(hidden, Wk, bk, qws, present, kws, vtws, 1);
  qkv_kernel<<<g1, 256, 0, stream>>>(hidden, Wv, bv, qws, present, kws, vtws, 2);

  dim3 g2(NS_ / 128, NB * NH);                         // (16, 24)
  attn_kernel<<<g2, 256, 0, stream>>>(qws, kws, vtws, ctx);
}
